// SelfAttention2d_2207613190593
// MI455X (gfx1250) — matrix operand order repaired and checked by reading, not compiled, not run on hardware
//
#include <hip/hip_runtime.h>
#include <hip/hip_bf16.h>

#define DEVI __device__ __forceinline__

typedef __bf16 bf16;
typedef __attribute__((ext_vector_type(16))) __bf16 bf16x16;
typedef __attribute__((ext_vector_type(8)))  __bf16 bf16x8;
typedef __attribute__((ext_vector_type(8)))  float  f32x8;
typedef __attribute__((ext_vector_type(4)))  unsigned int u32x4;
typedef __attribute__((ext_vector_type(8)))  int i32x8;
typedef __attribute__((ext_vector_type(4)))  int i32x4;

// ---------------- problem constants ----------------
constexpr int CB     = 16;
constexpr int CC     = 512;
constexpr int CH     = 64, CW = 64;
constexpr int HW     = CH * CW;          // 4096
constexpr int HEADS  = 8,  HD = 64;
constexpr int INNER  = HEADS * HD;       // 512
constexpr int WSZ    = 8;
constexpr int NWIN   = (CH / WSZ) * (CW / WSZ); // 64
constexpr int GROUPS = 32, CPG = CC / GROUPS;   // 16
constexpr int QKVROWS = 3 * INNER;       // 1536

// ---------------- LDS layout (bf16 element strides) ----------------
constexpr int XSTR = 520;                          // 512 + pad, per-pixel row
constexpr int QSTR = 72;                           // 64 + pad, per-token row
constexpr int VSTR = 72;
constexpr int XT_BYTES  = 64 * XSTR * 2;           // 66560 (>= Pt: 8*64*64*2)
constexpr int QH_ELEMS  = 64 * QSTR;
constexpr int QT_BYTES  = HEADS * QH_ELEMS * 2;    // 73728
constexpr int VH_ELEMS  = 64 * VSTR;
constexpr int VT_BYTES  = HEADS * VH_ELEMS * 2;    // 73728
constexpr int LDS_ATTN  = XT_BYTES + 2 * QT_BYTES + VT_BYTES;   // 287744 < 320KB

constexpr int OSTR   = 520;                        // 512 + 8 bf16 (16B pad/row)
constexpr int LDS_PROJ = 64 * OSTR * 2;            // 66560

// ---------------- WMMA helpers ----------------
DEVI f32x8 zero8() { f32x8 z = {0.f,0.f,0.f,0.f,0.f,0.f,0.f,0.f}; return z; }

// A operand, 16x32 bf16: lane<16 -> K 0..7 & 16..23 ; lane>=16 -> K 8..15 & 24..31
DEVI bf16x16 ld_a16(const bf16* p, int a_off /*0 or 8*/) {
    const bf16* q = p + a_off;
    bf16x8 lo = *(const bf16x8*)(q);
    bf16x8 hh = *(const bf16x8*)(q + 16);
    return __builtin_shufflevector(lo, hh, 0,1,2,3,4,5,6,7,8,9,10,11,12,13,14,15);
}
// B operand, 32x16 bf16: lane<16 -> K 0..7 & 16..23 ; lane>=16 -> K 8..15 & 24..31
DEVI bf16x16 ld_b16(const bf16* p, int b_off /*0 or 8*/) {
    const bf16* q = p + b_off;
    bf16x8 lo = *(const bf16x8*)(q);
    bf16x8 hh = *(const bf16x8*)(q + 16);
    return __builtin_shufflevector(lo, hh, 0,1,2,3,4,5,6,7,8,9,10,11,12,13,14,15);
}
DEVI f32x8 wmma_bf16(bf16x16 a, bf16x16 b, f32x8 c) {
    return __builtin_amdgcn_wmma_f32_16x16x32_bf16(false, a, false, b,
                                                   (short)0, c, false, false);
}

// CDNA5 async copy: global -> LDS through the async engine (ASYNCcnt),
// no VGPR round-trip. LDS address = low 32 bits of the generic pointer
// (flat aperture truncation rule).
DEVI void async_g2l_b128(void* lds_ptr, const void* gptr) {
    const unsigned lds_off = (unsigned)(uintptr_t)lds_ptr;
    asm volatile("global_load_async_to_lds_b128 %0, %1, off"
                 :: "v"(lds_off), "v"(gptr) : "memory");
}
DEVI void wait_asynccnt0() {
    asm volatile("s_wait_asynccnt 0x0" ::: "memory");
}

// CDNA5 Tensor Data Mover: one descriptor moves a contiguous run of
// `elems` bf16 from global to LDS, inserting 4 DWORDs of LDS padding
// after every 256 DWORDs (i.e. 16B pad per 1024B row -> OSTR layout).
DEVI void tdm_load_rows(void* lds_ptr, const void* gptr, unsigned elems) {
    const unsigned lds_base = (unsigned)(uintptr_t)lds_ptr;
    const unsigned long long ga = (unsigned long long)(uintptr_t)gptr;
    u32x4 g0;
    g0.x = 1u;                                           // count=1 (valid user D#)
    g0.y = lds_base;                                     // lds_addr
    g0.z = (unsigned)(ga & 0xffffffffu);                 // global_addr[31:0]
    g0.w = (unsigned)((ga >> 32) & 0x01ffffffu)          // global_addr[56:32]
         | (2u << 30);                                   // type = 2 ("image")
    i32x8 g1;
    g1[0] = (1 << 16)                                    // data_size = 2 bytes
          | (1 << 20)                                    // pad_enable
          | (7 << 22)                                    // pad_interval: 256 DWORDs
          | (3 << 25);                                   // pad_amount: 4 DWORDs
    g1[1] = (int)(0x8000u << 16);                        // tensor_dim0 = 32768 (lo16)
    g1[2] = (int)(1u << 16);                             // dim0 hi = 0; tensor_dim1 = 1
    g1[3] = (int)((elems & 0xffffu) << 16);              // dim1 hi = 0; tile_dim0
    g1[4] = 1;                                           // tile_dim1 = 1; tile_dim2 = 0
    g1[5] = 32768;                                       // tensor_dim0_stride lo
    g1[6] = 0;
    g1[7] = 0;
    const i32x4 gz4 = {0, 0, 0, 0};
    const i32x8 gz8 = {0, 0, 0, 0, 0, 0, 0, 0};
    // 6-arg prototype on this toolchain: (g0, g1, g2, g3, g4, cpol)
    __builtin_amdgcn_tensor_load_to_lds(g0, g1, gz4, gz4, gz8, 0);
}

// ==================================================================
// Kernel 1: GroupNorm statistics: mean / rstd per (batch, group)
// ==================================================================
__global__ void __launch_bounds__(256)
gn_stats_kernel(const float* __restrict__ x, float* __restrict__ stats)
{
    const int bg = blockIdx.x;                 // 0..511
    const int b  = bg >> 5;
    const int g  = bg & 31;
    const float* base = x + ((size_t)b * CC + (size_t)g * CPG) * HW;
    const int N = CPG * HW;                    // 65536

    float s = 0.f, ss = 0.f;
    for (int i = threadIdx.x; i < N; i += blockDim.x) {
        float v = base[i];
        s += v; ss += v * v;
    }
    #pragma unroll
    for (int off = 16; off > 0; off >>= 1) {
        s  += __shfl_down(s,  off);
        ss += __shfl_down(ss, off);
    }
    __shared__ float rs[8], rss[8];
    const int wid = threadIdx.x >> 5, lid = threadIdx.x & 31;
    if (lid == 0) { rs[wid] = s; rss[wid] = ss; }
    __syncthreads();
    if (threadIdx.x == 0) {
        float S = 0.f, SS = 0.f;
        #pragma unroll
        for (int i = 0; i < 8; ++i) { S += rs[i]; SS += rss[i]; }
        const float inv = 1.0f / (float)N;
        const float mu  = S * inv;
        const float var = SS * inv - mu * mu;
        stats[2*bg]   = mu;
        stats[2*bg+1] = rsqrtf(var + 1e-6f);
    }
}

// ==================================================================
// Kernel 2: one-time fp32 -> bf16 weight conversion
// ==================================================================
__global__ void __launch_bounds__(256)
cvt_kernel(const float* __restrict__ wqkv, const float* __restrict__ wout,
           bf16* __restrict__ wq_b, bf16* __restrict__ wo_b)
{
    const int i = blockIdx.x * blockDim.x + threadIdx.x;
    if (i < QKVROWS * CC) wq_b[i] = (bf16)wqkv[i];
    if (i < CC * INNER)   wo_b[i] = (bf16)wout[i];
}

// ==================================================================
// Kernel 3: fused GroupNorm-apply + QKV GEMM + window attention
// one workgroup (8 wave32 = 8 heads) per (batch, window)
// o_img output layout: [b*NWIN][64 tokens][INNER] bf16 (token-major)
// ==================================================================
__global__ void __launch_bounds__(256)
attn_kernel(const float* __restrict__ x, const float* __restrict__ stats,
            const float* __restrict__ gamma, const float* __restrict__ beta,
            const bf16* __restrict__ wq, bf16* __restrict__ o_img)
{
    extern __shared__ char smem[];
    bf16* Xt = (bf16*)smem;                              // [64 tok][XSTR ch]
    bf16* Qt = (bf16*)(smem + XT_BYTES);                 // [head][64 tok][QSTR d]
    bf16* Kt = (bf16*)(smem + XT_BYTES + QT_BYTES);      // [head][64 tok][QSTR d]
    bf16* Vd = (bf16*)(smem + XT_BYTES + 2 * QT_BYTES);  // [head][64 d][VSTR tok]
    bf16* Pt = Xt;                                       // reuse: [head][64 q][64 k]

    const int blk = blockIdx.x;
    const int b   = blk >> 6;
    const int win = blk & 63;
    const int wy0 = (win >> 3) * WSZ;
    const int wx0 = (win & 7) * WSZ;

    const float* xb = x + (size_t)b * CC * HW;

    // ---- stage normalized activation window, token-major, bf16 ----
    // pixel-fastest mapping: 8 consecutive lanes read 32B-contiguous global.
    for (int i = threadIdx.x; i < CC * 64; i += blockDim.x) {
        const int p = i & 63;
        const int c = i >> 6;
        const int yy = wy0 + (p >> 3);
        const int xx = wx0 + (p & 7);
        float v = xb[(size_t)c * HW + yy * CW + xx];
        const int sg = b * GROUPS + (c >> 4);
        const float mu = stats[2*sg], rsd = stats[2*sg+1];
        v = (v - mu) * rsd * gamma[c] + beta[c];
        Xt[p * XSTR + c] = (bf16)v;
    }
    __syncthreads();

    const int wave = threadIdx.x >> 5;
    const int lane = threadIdx.x & 31;
    const int l15  = lane & 15;
    const int hi   = (lane >> 4) & 1;
    const int a_off = hi ? 8 : 0;
    const int b_off = hi ? 8 : 0;

    // ---- phase 1: QKV = Wqkv(1536x512) x Xn(512x64) ----
    for (int mt8 = 0; mt8 < QKVROWS / 16 / 8; ++mt8) {
        const int mtile = mt8 * 8 + wave;
        const int mrow  = mtile * 16;
        f32x8 acc[4];
        #pragma unroll
        for (int nt = 0; nt < 4; ++nt) acc[nt] = zero8();

        const bf16* arow = wq + (size_t)(mrow + l15) * CC;
        #pragma unroll 4
        for (int kk = 0; kk < CC / 32; ++kk) {
            if (kk + 1 < CC / 32) __builtin_prefetch(arow + (kk + 1) * 32, 0, 3);
            const bf16x16 A = ld_a16(arow + kk * 32, a_off);
            #pragma unroll
            for (int nt = 0; nt < 4; ++nt) {
                const bf16x16 Bv = ld_b16(Xt + (nt*16 + l15) * XSTR + kk * 32, b_off);
                acc[nt] = wmma_bf16(A, Bv, acc[nt]);
            }
        }
        // scatter D tiles into Q (token-major), K (token-major), V (d-major)
        #pragma unroll
        for (int nt = 0; nt < 4; ++nt) {
            const int t = nt * 16 + l15;
            #pragma unroll
            for (int r = 0; r < 8; ++r) {
                const int row = mrow + r + hi * 8;
                const bf16 v = (bf16)acc[nt][r];
                if (row < INNER) {
                    Qt[(row >> 6) * QH_ELEMS + t * QSTR + (row & 63)] = v;
                } else if (row < 2 * INNER) {
                    const int rr = row - INNER;
                    Kt[(rr >> 6) * QH_ELEMS + t * QSTR + (rr & 63)] = v;
                } else {
                    const int rr = row - 2 * INNER;
                    Vd[(rr >> 6) * VH_ELEMS + (rr & 63) * VSTR + t] = v;
                }
            }
        }
    }
    __syncthreads();

    // ---- phase 2: attention, one head per wave ----
    const int h = wave;
    const bf16* Qh = Qt + h * QH_ELEMS;
    const bf16* Kh = Kt + h * QH_ELEMS;
    const bf16* Vh = Vd + h * VH_ELEMS;
    bf16*       Ph = Pt + h * 4096;

    f32x8 s[4][4];
    #pragma unroll
    for (int mt = 0; mt < 4; ++mt)
        #pragma unroll
        for (int nt = 0; nt < 4; ++nt) s[mt][nt] = zero8();

    #pragma unroll
    for (int kk = 0; kk < 2; ++kk) {              // K = hd = 64 -> 2 chunks
        bf16x16 Bk[4];
        #pragma unroll
        for (int nt = 0; nt < 4; ++nt)
            Bk[nt] = ld_b16(Kh + (nt*16 + l15) * QSTR + kk * 32, b_off);
        #pragma unroll
        for (int mt = 0; mt < 4; ++mt) {
            const bf16x16 A = ld_a16(Qh + (mt*16 + l15) * QSTR + kk * 32, a_off);
            #pragma unroll
            for (int nt = 0; nt < 4; ++nt)
                s[mt][nt] = wmma_bf16(A, Bk[nt], s[mt][nt]);
        }
    }

    // in-register softmax (rows live in one 16-lane half per the D layout)
    constexpr float scale = 0.125f;               // hd^-0.5
    #pragma unroll
    for (int mt = 0; mt < 4; ++mt) {
        #pragma unroll
        for (int r = 0; r < 8; ++r) {
            float m = -3.0e38f;
            #pragma unroll
            for (int nt = 0; nt < 4; ++nt) m = fmaxf(m, s[mt][nt][r]);
            #pragma unroll
            for (int off = 8; off > 0; off >>= 1) m = fmaxf(m, __shfl_xor(m, off));
            float sum = 0.f;
            #pragma unroll
            for (int nt = 0; nt < 4; ++nt) {
                const float e = __expf((s[mt][nt][r] - m) * scale);
                s[mt][nt][r] = e;
                sum += e;
            }
            #pragma unroll
            for (int off = 8; off > 0; off >>= 1) sum += __shfl_xor(sum, off);
            const float inv = 1.0f / sum;
            const int q = mt * 16 + r + hi * 8;
            #pragma unroll
            for (int nt = 0; nt < 4; ++nt)
                Ph[q * 64 + nt * 16 + l15] = (bf16)(s[mt][nt][r] * inv);
        }
    }
    // DS ops are in-order per wave; Ph is wave-private -> no barrier needed.

    // O = P(64x64) x V(64x64)
    f32x8 o[4][4];
    #pragma unroll
    for (int mt = 0; mt < 4; ++mt)
        #pragma unroll
        for (int nt = 0; nt < 4; ++nt) o[mt][nt] = zero8();

    #pragma unroll
    for (int kk = 0; kk < 2; ++kk) {
        bf16x16 Bv[4];
        #pragma unroll
        for (int nt = 0; nt < 4; ++nt)
            Bv[nt] = ld_b16(Vh + (nt*16 + l15) * VSTR + kk * 32, b_off);
        #pragma unroll
        for (int mt = 0; mt < 4; ++mt) {
            const bf16x16 A = ld_a16(Ph + (mt*16 + l15) * 64 + kk * 32, a_off);
            #pragma unroll
            for (int nt = 0; nt < 4; ++nt)
                o[mt][nt] = wmma_bf16(A, Bv[nt], o[mt][nt]);
        }
    }

    // write O token-major: [blk][tok][c]; lanes 0..15 hit 32B-contiguous bf16
    bf16* ob = o_img + (size_t)blk * 64 * INNER;
    #pragma unroll
    for (int mt = 0; mt < 4; ++mt) {
        #pragma unroll
        for (int r = 0; r < 8; ++r) {
            const int q = mt * 16 + r + hi * 8;
            #pragma unroll
            for (int nt = 0; nt < 4; ++nt) {
                const int c = h * HD + nt * 16 + l15;
                ob[q * INNER + c] = (bf16)o[mt][nt][r];
            }
        }
    }
}

// ==================================================================
// Kernel 4: out-projection + bias + residual
// one workgroup per (batch, window); O tile is one contiguous 64KB
// block staged half by TDM (TENSORcnt), half by async loads (ASYNCcnt)
// ==================================================================
__global__ void __launch_bounds__(256)
proj_kernel(const bf16* __restrict__ o_img, const bf16* __restrict__ wo,
            const float* __restrict__ b_out, const float* __restrict__ x,
            float* __restrict__ out)
{
    extern __shared__ char smem[];
    bf16* Ot = (bf16*)smem;                      // [64 tok][OSTR ch]

    const int blk = blockIdx.x;
    const int b   = blk >> 6;
    const int win = blk & 63;
    const int wy0 = (win >> 3) * WSZ;
    const int wx0 = (win & 7) * WSZ;

    const int wave = threadIdx.x >> 5;
    const bf16* ob = o_img + (size_t)blk * 64 * INNER;   // contiguous 64KB

    // tokens 0..31: single TDM descriptor (padding folded into the D#)
    if (wave == 0) {
        tdm_load_rows(Ot, ob, 32u * INNER);
    }
    // tokens 32..63: async global->LDS b128 copies from all waves
    for (int i = threadIdx.x; i < 32 * (INNER / 8); i += blockDim.x) {
        const int p  = 32 + (i >> 6);
        const int ck = i & 63;
        async_g2l_b128(Ot + p * OSTR + ck * 8, ob + p * INNER + ck * 8);
    }
    if (wave == 0) __builtin_amdgcn_s_wait_tensorcnt(0);
    wait_asynccnt0();
    __syncthreads();

    const int lane = threadIdx.x & 31;
    const int l15  = lane & 15;
    const int hi   = (lane >> 4) & 1;
    const int a_off = hi ? 8 : 0;
    const int b_off = hi ? 8 : 0;

    f32x8 acc[4][4];
    #pragma unroll
    for (int mt = 0; mt < 4; ++mt)
        #pragma unroll
        for (int nt = 0; nt < 4; ++nt) acc[mt][nt] = zero8();

    const int mbase = wave * 64;
    #pragma unroll 2
    for (int kk = 0; kk < INNER / 32; ++kk) {
        bf16x16 Bv[4];
        #pragma unroll
        for (int nt = 0; nt < 4; ++nt)
            Bv[nt] = ld_b16(Ot + (nt*16 + l15) * OSTR + kk * 32, b_off);
        #pragma unroll
        for (int mt = 0; mt < 4; ++mt) {
            const bf16* arow = wo + (size_t)(mbase + mt*16 + l15) * INNER + kk * 32;
            if (kk + 1 < INNER / 32) __builtin_prefetch(arow + 32, 0, 3);
            const bf16x16 A = ld_a16(arow, a_off);
            #pragma unroll
            for (int nt = 0; nt < 4; ++nt)
                acc[mt][nt] = wmma_bf16(A, Bv[nt], acc[mt][nt]);
        }
    }

    const float* xb = x   + (size_t)b * CC * HW;
    float*       yb = out + (size_t)b * CC * HW;
    #pragma unroll
    for (int mt = 0; mt < 4; ++mt) {
        #pragma unroll
        for (int r = 0; r < 8; ++r) {
            const int c  = mbase + mt * 16 + r + hi * 8;
            const float bo = b_out[c];
            #pragma unroll
            for (int nt = 0; nt < 4; ++nt) {
                const int tok = nt * 16 + l15;
                const int pix = (wy0 + (tok >> 3)) * CW + wx0 + (tok & 7);
                yb[(size_t)c * HW + pix] = acc[mt][nt][r] + bo + xb[(size_t)c * HW + pix];
            }
        }
    }
}

// ==================================================================
extern "C" void kernel_launch(void* const* d_in, const int* in_sizes, int n_in,
                              void* d_out, int out_size, void* d_ws, size_t ws_size,
                              hipStream_t stream)
{
    const float* x     = (const float*)d_in[0];
    const float* gamma = (const float*)d_in[1];
    const float* beta  = (const float*)d_in[2];
    const float* wqkv  = (const float*)d_in[3];
    const float* wout  = (const float*)d_in[4];
    const float* bout  = (const float*)d_in[5];
    float* out = (float*)d_out;

    char* ws = (char*)d_ws;
    float* stats = (float*)ws;                 ws += 4096;                       // 512*2 f32
    bf16*  wq_b  = (bf16*)ws;                  ws += (size_t)QKVROWS * CC * 2;   // 1.5 MB
    bf16*  wo_b  = (bf16*)ws;                  ws += (size_t)CC * INNER * 2;     // 0.5 MB
    bf16*  o_img = (bf16*)ws;                  // [B*NWIN][64][INNER] bf16 = 64 MB

    gn_stats_kernel<<<dim3(CB * GROUPS), dim3(256), 0, stream>>>(x, stats);
    cvt_kernel<<<dim3((QKVROWS * CC + 255) / 256), dim3(256), 0, stream>>>(
        wqkv, wout, wq_b, wo_b);
    attn_kernel<<<dim3(CB * NWIN), dim3(256), LDS_ATTN, stream>>>(
        x, stats, gamma, beta, wq_b, o_img);
    proj_kernel<<<dim3(CB * NWIN), dim3(256), LDS_PROJ, stream>>>(
        o_img, wo_b, bout, x, out);

    (void)in_sizes; (void)n_in; (void)out_size; (void)ws_size;
}
